// VQ_AENB_18631568130295
// MI455X (gfx1250) — compile-verified, matrix operand order checked by reading, not run on hardware
//
#include <hip/hip_runtime.h>
#include <hip/hip_bf16.h>

// ---------------------------------------------------------------------------
// VQ-VAE forward on MI455X (gfx1250, wave32).
// All GEMMs use v_wmma_f32_16x16x32_bf16 (bf16 in, f32 accum).
// Software-pipelined: register staging + double-buffered LDS (1 barrier/iter).
// LDS staging uses the WMMA fragment layout from CDNA5 ISA 7.12.2 so each
// lane fetches its 16-bf16 fragment with two ds_load_b128.
// ---------------------------------------------------------------------------

typedef __attribute__((ext_vector_type(16))) __bf16 v16bf;
typedef __attribute__((ext_vector_type(8)))  float  v8f;

#define BM 128
#define BN 128
#define BK 32
#define THREADS 256   // 8 wave32 waves; wave grid 4 (M) x 2 (N); wave tile 32x64

enum { ACT_NONE = 0, ACT_RELU = 1, ACT_VQVAE_OUT = 2 };

__device__ __forceinline__ unsigned short f32_bf16_rne(float f) {
  unsigned int u = __float_as_uint(f);
  unsigned int r = u + 0x7FFFu + ((u >> 16) & 1u);   // round-to-nearest-even
  return (unsigned short)(r >> 16);
}

// A fragment layout (16-bit A 16x32, ISA 7.12.2):
//   lane = (m&15) + 16*hi  where hi = (k>>3)&1
//   elem = (k&7) + (k>=16 ? 8 : 0)
// B fragment (32x16 K-major) uses the mirrored mapping with n in place of m.
__device__ __forceinline__ int frag_lane(int mn, int k) {
  return (mn & 15) + (((k >> 3) & 1) << 4);
}
__device__ __forceinline__ int frag_elem(int k) {
  return (k & 7) + ((k >> 4) << 3);
}

template <int ACT>
__global__ __launch_bounds__(THREADS) void gemm_bf16_wmma(
    const float* __restrict__ A,     // [M,K] row-major fp32
    const float* __restrict__ Bm,    // [K,N] row-major fp32
    const float* __restrict__ bias,  // [N]
    float* __restrict__ C,           // [M,N] (or split mu/theta for VQVAE_OUT)
    int M, int N, int K,
    int G,                           // split column for VQVAE_OUT epilogue
    unsigned long long outHalf)      // element offset of theta block
{
  // double-buffered fragment-major tiles: 2 * (8KB + 8KB) = 32 KB
  __shared__ unsigned short lds_a[2][BM * BK];
  __shared__ unsigned short lds_b[2][BN * BK];

  const int tid    = threadIdx.x;
  const int lane   = tid & 31;
  const int wid    = tid >> 5;
  const int wave_m = wid & 3;        // 0..3 -> 32-row slab
  const int wave_n = wid >> 2;       // 0..1 -> 64-col slab
  const int block_row = blockIdx.y * BM;
  const int block_col = blockIdx.x * BN;

  v8f acc[2][4];
#pragma unroll
  for (int mi = 0; mi < 2; ++mi)
#pragma unroll
    for (int ni = 0; ni < 4; ++ni) acc[mi][ni] = v8f{};

  const int kTiles = K / BK;   // every K in this model is a multiple of 32

  // ---- per-thread load pattern (loop-invariant):
  //   A: rows m0+8i (i=0..15), fixed col a_k   -> base += BK per tile
  //   B: rows b_k0+2i (i=0..15), fixed col b_n -> base += BK*N per tile
  const int a_m0 = tid >> 5;                 // 0..7
  const int a_k  = tid & 31;                 // 0..31
  const int b_k0 = tid >> 7;                 // 0..1
  const int b_n  = tid & 127;                // 0..127
  const bool b_ok = (block_col + b_n) < N;   // zero-pad past N (invariant)
  const int  b_nc = b_ok ? (block_col + b_n) : (N - 1);

  const float* aPtr = A + (size_t)(block_row + a_m0) * K + a_k;
  const float* bPtr = Bm + (size_t)b_k0 * N + b_nc;
  const size_t aStride = (size_t)8 * K;      // between my A rows
  const size_t bStride = (size_t)2 * N;      // between my B rows
  const size_t bTile   = (size_t)BK * N;     // B advance per K-tile

  float aReg[16], bReg[16];

  auto load_stage = [&]() {
#pragma unroll
    for (int i = 0; i < 16; ++i) aReg[i] = aPtr[i * aStride];
#pragma unroll
    for (int i = 0; i < 16; ++i) bReg[i] = b_ok ? bPtr[i * bStride] : 0.0f;
    aPtr += BK;
    bPtr += bTile;
  };
  auto store_stage = [&](int buf) {
#pragma unroll
    for (int i = 0; i < 16; ++i) {
      int m = a_m0 + 8 * i;
      lds_a[buf][(m >> 4) * (32 * 16) + frag_lane(m, a_k) * 16 + frag_elem(a_k)] =
          f32_bf16_rne(aReg[i]);
    }
#pragma unroll
    for (int i = 0; i < 16; ++i) {
      int k = b_k0 + 2 * i;
      lds_b[buf][(b_n >> 4) * (32 * 16) + frag_lane(b_n, k) * 16 + frag_elem(k)] =
          f32_bf16_rne(bReg[i]);
    }
  };

  // prologue: stage tile 0 into buffer 0
  load_stage();
  store_stage(0);
  __syncthreads();

  for (int kt = 0; kt < kTiles; ++kt) {
    const int cur = kt & 1;
    const int nxt = cur ^ 1;

    // issue global loads for tile kt+1 while WMMAs chew on tile kt
    if (kt + 1 < kTiles) {
      load_stage();
      // pull tile kt+2 toward the caches (global_prefetch_b8)
      __builtin_prefetch(aPtr + (size_t)(tid & 7) * aStride, 0, 1);
      __builtin_prefetch(bPtr + (size_t)(tid & 1) * bStride, 0, 1);
    }

    // fragment loads (2x ds_load_b128 each) + 8 WMMAs per wave
    const v16bf* pa = reinterpret_cast<const v16bf*>(lds_a[cur]);
    const v16bf* pb = reinterpret_cast<const v16bf*>(lds_b[cur]);
    v16bf af[2], bfr[4];
#pragma unroll
    for (int mi = 0; mi < 2; ++mi) af[mi] = pa[(wave_m * 2 + mi) * 32 + lane];
#pragma unroll
    for (int ni = 0; ni < 4; ++ni) bfr[ni] = pb[(wave_n * 4 + ni) * 32 + lane];

#pragma unroll
    for (int mi = 0; mi < 2; ++mi)
#pragma unroll
      for (int ni = 0; ni < 4; ++ni)
        acc[mi][ni] = __builtin_amdgcn_wmma_f32_16x16x32_bf16(
            false, af[mi], false, bfr[ni], (short)0, acc[mi][ni], false, false);

    // stores for tile kt+1 go to the other buffer -> no WAR with readers
    if (kt + 1 < kTiles) store_stage(nxt);
    __syncthreads();
  }

  // ---- epilogue: C/D layout (ISA 7.12.2): lane&15 = N, lane>>4 picks M 0-7/8-15
#pragma unroll
  for (int mi = 0; mi < 2; ++mi) {
#pragma unroll
    for (int ni = 0; ni < 4; ++ni) {
      int col  = block_col + wave_n * 64 + ni * 16 + (lane & 15);
      int row0 = block_row + wave_m * 32 + mi * 16 + ((lane >> 4) << 3);
      if (col < N) {
        float bv = bias[col];
#pragma unroll
        for (int r = 0; r < 8; ++r) {
          float v = acc[mi][ni][r] + bv;
          int row = row0 + r;
          if (ACT == ACT_RELU) {
            C[(size_t)row * N + col] = fmaxf(v, 0.0f);
          } else if (ACT == ACT_NONE) {
            C[(size_t)row * N + col] = v;
          } else {  // ACT_VQVAE_OUT: mu = clip(exp(.)), theta = clip(softplus(.))
            if (col < G) {
              float mu = __expf(v);
              C[(size_t)row * G + col] = fminf(fmaxf(mu, 1e-6f), 1e6f);
            } else {
              float sp = (v > 20.0f) ? v : __logf(1.0f + __expf(v));
              C[outHalf + (size_t)row * G + (col - G)] =
                  fminf(fmaxf(sp, 1e-4f), 1e4f);
            }
          }
        }
      }
    }
  }
}

// ---------------------------------------------------------------------------
// VQ nearest-code search: one wave per batch row. z row = 4 floats/lane.
// Codebook (256 KB) stays hot in L2/WGP$. Accumulates sum((z-z_q)^2).
// ---------------------------------------------------------------------------
__global__ __launch_bounds__(256) void vq_nearest(
    const float* __restrict__ z, const float* __restrict__ cb,
    float* __restrict__ zq, float* __restrict__ lossAcc) {
  const int lane = threadIdx.x & 31;
  const int wid  = threadIdx.x >> 5;
  const int b    = blockIdx.x * 8 + wid;   // 8 waves per block

  float zr[4];
#pragma unroll
  for (int j = 0; j < 4; ++j) zr[j] = z[(size_t)b * 128 + lane + 32 * j];

  float best = 3.4e38f;
  int bestk = 0;
  for (int k = 0; k < 512; ++k) {
    float d = 0.0f;
#pragma unroll
    for (int j = 0; j < 4; ++j) {
      float c = cb[(size_t)k * 128 + lane + 32 * j];
      float t = zr[j] - c;
      d = fmaf(t, t, d);
    }
    // wave32 butterfly reduce -> every lane holds the full distance
#pragma unroll
    for (int off = 16; off > 0; off >>= 1) d += __shfl_xor(d, off, 32);
    if (d < best) { best = d; bestk = k; }
  }

  float ls = 0.0f;
#pragma unroll
  for (int j = 0; j < 4; ++j) {
    float c = cb[(size_t)bestk * 128 + lane + 32 * j];
    zq[(size_t)b * 128 + lane + 32 * j] = c;
    float t = zr[j] - c;
    ls = fmaf(t, t, ls);
  }
#pragma unroll
  for (int off = 16; off > 0; off >>= 1) ls += __shfl_xor(ls, off, 32);
  if (lane == 0) atomicAdd(lossAcc, ls);
}

__global__ void vq_loss_init(float* lossAcc) { *lossAcc = 0.0f; }

__global__ void vq_loss_finalize(const float* lossAcc, float* out,
                                 unsigned long long idx, float scale) {
  out[idx] = scale * (*lossAcc);   // (1 + 0.25) * mean((z - z_q)^2)
}

// ---------------------------------------------------------------------------
extern "C" void kernel_launch(void* const* d_in, const int* in_sizes, int n_in,
                              void* d_out, int out_size, void* d_ws, size_t ws_size,
                              hipStream_t stream) {
  const float* x    = (const float*)d_in[0];
  const float* We1  = (const float*)d_in[1];
  const float* be1  = (const float*)d_in[2];
  const float* We2  = (const float*)d_in[3];
  const float* be2  = (const float*)d_in[4];
  const float* We3  = (const float*)d_in[5];
  const float* be3  = (const float*)d_in[6];
  const float* cb   = (const float*)d_in[7];
  const float* Wd1  = (const float*)d_in[8];
  const float* bd1  = (const float*)d_in[9];
  const float* Wd2  = (const float*)d_in[10];
  const float* bd2  = (const float*)d_in[11];
  const float* Wd3  = (const float*)d_in[12];
  const float* bd3  = (const float*)d_in[13];

  const int Bb = 4096, G = 20000, H1 = 1024, H2 = 512, L = 128;

  // workspace layout (~28 MB): h1 | h2 | z | zq | loss
  float* h1   = (float*)d_ws;                 // 4096*1024
  float* h2   = h1 + (size_t)Bb * H1;         // 4096*512 (reused by decoder)
  float* z    = h2 + (size_t)Bb * H2;         // 4096*128
  float* zq   = z  + (size_t)Bb * L;          // 4096*128
  float* loss = zq + (size_t)Bb * L;          // 1

  dim3 blk(THREADS);
  const unsigned long long BG = (unsigned long long)Bb * G;

  vq_loss_init<<<1, 1, 0, stream>>>(loss);

  // encoder: x -> h1 -> h2 -> z
  gemm_bf16_wmma<ACT_RELU><<<dim3(H1 / BN, Bb / BM), blk, 0, stream>>>(
      x, We1, be1, h1, Bb, H1, G, 0, 0ull);
  gemm_bf16_wmma<ACT_RELU><<<dim3(H2 / BN, Bb / BM), blk, 0, stream>>>(
      h1, We2, be2, h2, Bb, H2, H1, 0, 0ull);
  gemm_bf16_wmma<ACT_NONE><<<dim3(1, Bb / BM), blk, 0, stream>>>(
      h2, We3, be3, z, Bb, L, H2, 0, 0ull);

  // vector quantization (+ commitment/codebook loss accumulation)
  vq_nearest<<<Bb / 8, 256, 0, stream>>>(z, cb, zq, loss);

  // decoder (straight-through forward uses z_q): zq -> h2 -> h1 -> d_out
  gemm_bf16_wmma<ACT_RELU><<<dim3(H2 / BN, Bb / BM), blk, 0, stream>>>(
      zq, Wd1, bd1, h2, Bb, H2, L, 0, 0ull);
  gemm_bf16_wmma<ACT_RELU><<<dim3(H1 / BN, Bb / BM), blk, 0, stream>>>(
      h2, Wd2, bd2, h1, Bb, H1, H2, 0, 0ull);
  gemm_bf16_wmma<ACT_VQVAE_OUT>
      <<<dim3((2 * G + BN - 1) / BN, Bb / BM), blk, 0, stream>>>(
          h1, Wd3, bd3, (float*)d_out, Bb, 2 * G, H1, G, BG);

  vq_loss_finalize<<<1, 1, 0, stream>>>(
      loss, (float*)d_out, 2ull * BG, 1.25f / ((float)Bb * (float)L));
}
